// DCMHAttention_45131516346466
// MI455X (gfx1250) — matrix operand order repaired and checked by reading, not compiled, not run on hardware
//
#include <hip/hip_runtime.h>
#include <stdint.h>

// ---------------- problem constants (match reference) ----------------
#define B_   2
#define T_   1024
#define S_   1024
#define D_   2048
#define NH   16
#define HD_  128
#define KDYN 128     // dyn hidden per c
#define C4   4
#define DHD  2
#define QK_SCALE 0.08838834764831845f  // HD^-0.5

typedef __attribute__((ext_vector_type(16))) __bf16 v16bf;
typedef __attribute__((ext_vector_type(8)))  float  v8f;
typedef __attribute__((ext_vector_type(4)))  int    i4;

union FragBF { v16bf v; uint4 q[2]; };

#ifndef __has_builtin
#define __has_builtin(x) 0
#endif
#if __has_builtin(__builtin_amdgcn_global_load_async_to_lds_b128)
#define HAVE_ASYNC_LDS 1
#else
#define HAVE_ASYNC_LDS 0
#endif

#define AS1 __attribute__((address_space(1)))
#define AS3 __attribute__((address_space(3)))

// 16B global -> LDS copy; async (ASYNCcnt) when the toolchain exposes it.
__device__ __forceinline__ void cp16(const void* g, void* l) {
#if HAVE_ASYNC_LDS
    __builtin_amdgcn_global_load_async_to_lds_b128(
        (AS1 i4*)(i4*)(void*)g, (AS3 i4*)(i4*)l, /*offset=*/0, /*cpol=*/0);
#else
    *(uint4*)l = *(const uint4*)g;
#endif
}
__device__ __forceinline__ void wait_async0() {
#if HAVE_ASYNC_LDS
#if __has_builtin(__builtin_amdgcn_s_wait_asynccnt)
    __builtin_amdgcn_s_wait_asynccnt(0);
#else
    asm volatile("s_wait_asynccnt 0x0" ::: "memory");
#endif
#endif
}

// round-to-nearest-even f32 -> bf16 (stored as u16)
__device__ __forceinline__ uint16_t f2bf(float f) {
    uint32_t u = __float_as_uint(f);
    u += 0x7FFFu + ((u >> 16) & 1u);
    return (uint16_t)(u >> 16);
}

// ======================================================================
// Batched bf16 WMMA GEMM:  C[M,N] = A[M,K] * B[K,N], B stored transposed
// as Bt[N,K].  Workgroup = 256 threads = 8 waves, tile 128(M) x 128(N),
// K-step 32, double-buffered async global->LDS staging.
// Each wave owns a 32(M) x 64(N) sub-tile = 2x4 WMMA tiles.
// ======================================================================
#define TM 128
#define TN 128
#define TK 32
#define LP 40   // LDS row pitch in bf16 elems (80B, 16B-aligned, pad vs banks)

// A-matrix 16x32 bf16 fragment per ISA table: lane half h ->
//   elems 0..7  = K 8h..8h+7,  elems 8..15 = K 16+8h..16+8h+7
__device__ __forceinline__ v16bf load_a_frag(const uint16_t* s, int row0, int lane) {
    const uint16_t* p = s + (size_t)(row0 + (lane & 15)) * LP;
    int h = lane >> 4;
    FragBF f;
    f.q[0] = *(const uint4*)(p + 8 * h);
    f.q[1] = *(const uint4*)(p + 16 + 8 * h);
    return f.v;
}
// B-matrix 32x16 bf16 fragment: lane half h -> elems 0..7 = K 8h..8h+7, elems 8..15 = K 16+8h..16+8h+7
__device__ __forceinline__ v16bf load_b_frag(const uint16_t* s, int col0, int lane) {
    const uint16_t* p = s + (size_t)(col0 + (lane & 15)) * LP;
    FragBF f;
    f.q[0] = *(const uint4*)(p + 8 * (lane >> 4));
    f.q[1] = *(const uint4*)(p + 16 + 8 * (lane >> 4));
    return f.v;
}

__device__ __forceinline__ v8f wmma_bf16(v16bf a, v16bf b, v8f c) {
    return __builtin_amdgcn_wmma_f32_16x16x32_bf16(
        /*neg_a=*/false, a, /*neg_b=*/false, b,
        /*c_mod=*/(short)0, c, /*reuse_a=*/false, /*reuse_b=*/false);
}

__global__ __launch_bounds__(256)
void gemm_bf16_wmma(const uint16_t* __restrict__ A, long long aO, long long aI, int lda,
                    const uint16_t* __restrict__ Bt, long long bO, long long bI, int ldb,
                    float* __restrict__ C, long long cO, long long cI, int ldc,
                    int M, int N, int K, int zInner)
{
    __shared__ uint16_t As[2][TM * LP];
    __shared__ uint16_t Bs[2][TN * LP];

    const int z  = blockIdx.z;
    const uint16_t* Ab = A  + (size_t)(z / zInner) * aO + (size_t)(z % zInner) * aI;
    const uint16_t* Bb = Bt + (size_t)(z / zInner) * bO + (size_t)(z % zInner) * bI;
    float*          Cb = C  + (size_t)(z / zInner) * cO + (size_t)(z % zInner) * cI;

    const int tid  = threadIdx.x;
    const int lane = tid & 31;
    const int wave = tid >> 5;
    const int wm   = wave & 3;    // 0..3 -> M sub-tile of 32
    const int wn   = wave >> 2;   // 0..1 -> N sub-tile of 64
    const int m0   = blockIdx.y * TM;
    const int n0   = blockIdx.x * TN;

    v8f acc[2][4];
    v8f zero = {0.f,0.f,0.f,0.f,0.f,0.f,0.f,0.f};
    #pragma unroll
    for (int i = 0; i < 2; ++i)
        #pragma unroll
        for (int j = 0; j < 4; ++j) acc[i][j] = zero;

    // staging: 128 rows x 32 elems each for A and B; 256 threads:
    // each thread moves one 32B row-half of A and one of B (2x b128 each).
    const int srow = tid >> 1, sseg = tid & 1;     // seg = 16 elems (32B)
    const uint16_t* gA = Ab + (size_t)(m0 + srow) * lda + sseg * 16;
    const uint16_t* gB = Bb + (size_t)(n0 + srow) * ldb + sseg * 16;
    const int lOff = srow * LP + sseg * 16;

    // preload K-tile 0 into buffer 0
    cp16(gA,     &As[0][lOff]);
    cp16(gA + 8, &As[0][lOff + 8]);
    cp16(gB,     &Bs[0][lOff]);
    cp16(gB + 8, &Bs[0][lOff + 8]);

    int buf = 0;
    for (int k0 = 0; k0 < K; k0 += TK) {
        wait_async0();       // my async stores into As/Bs[buf] landed
        __syncthreads();     // everyone's landed; everyone done reading buf^1

        if (k0 + TK < K) {   // stage next K-tile into the other buffer
            const uint16_t* nA = gA + k0 + TK;
            const uint16_t* nB = gB + k0 + TK;
            cp16(nA,     &As[buf ^ 1][lOff]);
            cp16(nA + 8, &As[buf ^ 1][lOff + 8]);
            cp16(nB,     &Bs[buf ^ 1][lOff]);
            cp16(nB + 8, &Bs[buf ^ 1][lOff + 8]);
            __builtin_prefetch(nA + TK, 0, 1);   // global_prefetch_b8 one tile ahead
            __builtin_prefetch(nB + TK, 0, 1);
        }

        v16bf af[2], bf[4];
        #pragma unroll
        for (int i = 0; i < 2; ++i) af[i] = load_a_frag(As[buf], wm * 32 + i * 16, lane);
        #pragma unroll
        for (int j = 0; j < 4; ++j) bf[j] = load_b_frag(Bs[buf], wn * 64 + j * 16, lane);
        #pragma unroll
        for (int i = 0; i < 2; ++i)
            #pragma unroll
            for (int j = 0; j < 4; ++j)
                acc[i][j] = wmma_bf16(af[i], bf[j], acc[i][j]);

        buf ^= 1;
    }

    // C/D layout: lane col = lane&15, row = vgpr + 8*(lane>>4)
    const int colb = n0 + wn * 64 + (lane & 15);
    const int rowb = m0 + wm * 32 + (lane >> 4) * 8;
    #pragma unroll
    for (int i = 0; i < 2; ++i)
        #pragma unroll
        for (int j = 0; j < 4; ++j)
            #pragma unroll
            for (int r = 0; r < 8; ++r)
                Cb[(size_t)(rowb + i * 16 + r) * ldc + colb + j * 16] = acc[i][j][r];
}

// ======================================================================
// Elementwise / small kernels
// ======================================================================
__global__ void f32_to_bf16_k(const float* __restrict__ in, uint16_t* __restrict__ out, int n) {
    int i = blockIdx.x * blockDim.x + threadIdx.x;
    if (i < n) out[i] = f2bf(in[i]);
}

// out[c*rows + r] = in[r*cols + c]   (bf16 convert+transpose)
__global__ void transpose_to_bf16_k(const float* __restrict__ in, uint16_t* __restrict__ out,
                                    int rows, int cols) {
    int i = blockIdx.x * blockDim.x + threadIdx.x;
    if (i >= rows * cols) return;
    int c = i / rows, r = i - c * rows;
    out[i] = f2bf(in[(size_t)r * cols + c]);
}

// qkv f32 [B,T,3D] -> q_bf[(b,n),T,HD] (scaled), k_bf[(b,n),S,HD], vT_bf[(b,n),HD,S]
__global__ void split_qkv_k(const float* __restrict__ qkv, uint16_t* __restrict__ q,
                            uint16_t* __restrict__ k, uint16_t* __restrict__ vt) {
    int i = blockIdx.x * blockDim.x + threadIdx.x;
    if (i >= B_ * T_ * D_) return;
    int d = i % D_, t = (i / D_) % T_, b = i / (D_ * T_);
    int n = d / HD_, h = d % HD_;
    const float* row = qkv + (size_t)(b * T_ + t) * (3 * D_);
    size_t bn = (size_t)(b * NH + n);
    q [bn * T_ * HD_ + (size_t)t * HD_ + h] = f2bf(row[d] * QK_SCALE);
    k [bn * T_ * HD_ + (size_t)t * HD_ + h] = f2bf(row[D_ + d]);
    vt[bn * HD_ * T_ + (size_t)h * T_ + t]  = f2bf(row[2 * D_ + d]);
}

__global__ void gelu_k(float* __restrict__ p, int n) {   // tanh approximation (jax default)
    int i = blockIdx.x * blockDim.x + threadIdx.x;
    if (i >= n) return;
    float x = p[i];
    float c = 0.7978845608028654f * (x + 0.044715f * x * x * x);
    p[i] = 0.5f * x * (1.f + tanhf(c));
}

// w = einsum('btck,ckim->btcim'); rmsnorm over n for the w1 (im<2) half.
// w_all layout: [bt][c][part][i][n], part0 = w1 (normed), part1 = w2.
__global__ __launch_bounds__(256)
void dyn_w_proj_k(const float* __restrict__ dwh, const float* __restrict__ qkw,
                  float* __restrict__ w_all) {
    int bt = blockIdx.x;
    __shared__ float h[C4 * KDYN];
    __shared__ float wv[256];
    for (int i = threadIdx.x; i < C4 * KDYN; i += 256) h[i] = dwh[(size_t)bt * (C4 * KDYN) + i];
    __syncthreads();
    int c = threadIdx.x >> 6, idx = threadIdx.x & 63, im = idx >> 4, n = idx & 15;
    float acc = 0.f;
    #pragma unroll 8
    for (int kk = 0; kk < KDYN; ++kk)
        acc += h[c * KDYN + kk] * qkw[((size_t)(c * KDYN + kk) * (2 * DHD) + im) * NH + n];
    wv[threadIdx.x] = acc;
    __syncthreads();
    if (im < DHD) {   // rmsnorm over last axis N (16 elems) at fixed (c, i)
        float ss = 0.f;
        #pragma unroll
        for (int j = 0; j < NH; ++j) { float x = wv[(c * 4 + im) * NH + j]; ss += x * x; }
        acc *= rsqrtf(ss * (1.f / NH) + 1e-6f);
    }
    int part = im >> 1, i2 = im & 1;
    w_all[(size_t)bt * 256 + (size_t)(c * 2 + part) * 32 + i2 * 16 + n] = acc;
}

// dd = tanh(x @ dd_w) : [B*T, 4N]
__global__ void dyn_dd_k(const float* __restrict__ x, const float* __restrict__ dd_w,
                         float* __restrict__ dd) {
    int i = blockIdx.x * blockDim.x + threadIdx.x;
    if (i >= B_ * T_ * 4 * NH) return;
    int m = i & 63, bt = i >> 6;
    float acc = 0.f;
    for (int d = 0; d < D_; ++d) acc += x[(size_t)bt * D_ + d] * dd_w[(size_t)d * 64 + m];
    dd[i] = tanhf(acc);
}

// Cross-head projection (rank-2 mixing across the 16 heads + diag terms),
// pointwise in (b,t,s).  PRE also applies the causal mask; POST also emits
// bf16 probs (A operand of the o-GEMM).
template <bool POST>
__global__ __launch_bounds__(256)
void cross_head_proj_k(float* __restrict__ io, const float* __restrict__ w_all,
                       const float* __restrict__ dd, uint16_t* __restrict__ probs_bf) {
    const int s = blockIdx.x * 256 + threadIdx.x;
    const int t = blockIdx.y, b = blockIdx.z;
    const int cq = POST ? 2 : 0, ck = POST ? 3 : 1;

    __shared__ float sQW[2][DHD][NH];   // [part][i][n]
    __shared__ float sQDD[NH];
    const float* wt = w_all + (size_t)(b * T_ + t) * 256;
    if (threadIdx.x < 64) {
        int part = threadIdx.x >> 5, i2 = (threadIdx.x >> 4) & 1, n = threadIdx.x & 15;
        sQW[part][i2][n] = wt[(size_t)(cq * 2 + part) * 32 + i2 * 16 + n];
    } else if (threadIdx.x < 80) {
        int n = threadIdx.x - 64;
        sQDD[n] = dd[(size_t)(b * T_ + t) * 64 + cq * 16 + n];
    }
    __syncthreads();

    const float* wk = w_all + (size_t)(b * T_ + s) * 256;
    float kw1[DHD][NH], kw2[DHD][NH], kdd[NH];
    #pragma unroll
    for (int i2 = 0; i2 < DHD; ++i2)
        #pragma unroll
        for (int n = 0; n < NH; ++n) {
            kw1[i2][n] = wk[(size_t)(ck * 2 + 0) * 32 + i2 * 16 + n];
            kw2[i2][n] = wk[(size_t)(ck * 2 + 1) * 32 + i2 * 16 + n];
        }
    #pragma unroll
    for (int n = 0; n < NH; ++n) kdd[n] = dd[(size_t)(b * T_ + s) * 64 + ck * 16 + n];

    float v[NH];
    #pragma unroll
    for (int n = 0; n < NH; ++n)
        v[n] = io[((size_t)(b * NH + n) * T_ + t) * S_ + s];

    float hq0 = 0.f, hq1 = 0.f, hk0 = 0.f, hk1 = 0.f;
    #pragma unroll
    for (int n = 0; n < NH; ++n) {
        hq0 += v[n] * sQW[0][0][n];
        hq1 += v[n] * sQW[0][1][n];
        hk0 += v[n] * kw1[0][n];
        hk1 += v[n] * kw1[1][n];
    }
    #pragma unroll
    for (int n = 0; n < NH; ++n) {
        float o = v[n]
                + hq0 * sQW[1][0][n] + hq1 * sQW[1][1][n]
                + hk0 * kw2[0][n]    + hk1 * kw2[1][n]
                + v[n] * (sQDD[n] + kdd[n]);
        if (!POST && s > t) o = -3.402823466e38f;   // causal mask after pre-proj
        size_t oi = ((size_t)(b * NH + n) * T_ + t) * S_ + s;
        io[oi] = o;
        if (POST) probs_bf[oi] = f2bf(o);
    }
}

// row softmax: one wave per row of length S_
__global__ __launch_bounds__(256)
void softmax_rows_k(float* __restrict__ io, int nRows) {
    int lane = threadIdx.x & 31, wave = threadIdx.x >> 5;
    int row = blockIdx.x * 8 + wave;
    if (row >= nRows) return;
    float* p = io + (size_t)row * S_;
    float m = -3.402823466e38f;
    for (int j = lane; j < S_; j += 32) m = fmaxf(m, p[j]);
    #pragma unroll
    for (int o = 16; o > 0; o >>= 1) m = fmaxf(m, __shfl_xor(m, o, 32));
    float sum = 0.f;
    for (int j = lane; j < S_; j += 32) { float e = __expf(p[j] - m); p[j] = e; sum += e; }
    #pragma unroll
    for (int o = 16; o > 0; o >>= 1) sum += __shfl_xor(sum, o, 32);
    float inv = 1.f / sum;
    for (int j = lane; j < S_; j += 32) p[j] *= inv;
}

// ======================================================================
// host-side orchestration
// ======================================================================
static inline void* ws_take(char*& cur, size_t bytes) {
    void* p = (void*)cur;
    cur += (bytes + 255) & ~(size_t)255;
    return p;
}

extern "C" void kernel_launch(void* const* d_in, const int* in_sizes, int n_in,
                              void* d_out, int out_size, void* d_ws, size_t ws_size,
                              hipStream_t stream) {
    (void)in_sizes; (void)n_in; (void)out_size; (void)ws_size;
    const float* x     = (const float*)d_in[0];   // [B,T,D]
    const float* w_qkv = (const float*)d_in[1];   // [D,3D]
    const float* w_o   = (const float*)d_in[2];   // [D,D]
    const float* dw1   = (const float*)d_in[3];   // [D, C4*KDYN]
    const float* qkw   = (const float*)d_in[4];   // [C4,KDYN,2*DHD,NH]
    const float* dd_w  = (const float*)d_in[5];   // [D, 4*NH]

    char* cur = (char*)d_ws;
    uint16_t* x_bf    = (uint16_t*)ws_take(cur, (size_t)B_*T_*D_*2);
    uint16_t* wqkvT   = (uint16_t*)ws_take(cur, (size_t)3*D_*D_*2);
    uint16_t* wo_bf   = (uint16_t*)ws_take(cur, (size_t)D_*D_*2);
    uint16_t* dw1T    = (uint16_t*)ws_take(cur, (size_t)C4*KDYN*D_*2);
    float*    qkvf    = (float*)   ws_take(cur, (size_t)B_*T_*3*D_*4);
    uint16_t* q_bf    = (uint16_t*)ws_take(cur, (size_t)B_*NH*T_*HD_*2);
    uint16_t* k_bf    = (uint16_t*)ws_take(cur, (size_t)B_*NH*T_*HD_*2);
    uint16_t* vT_bf   = (uint16_t*)ws_take(cur, (size_t)B_*NH*HD_*T_*2);
    float*    dwh     = (float*)   ws_take(cur, (size_t)B_*T_*C4*KDYN*4);
    float*    w_all   = (float*)   ws_take(cur, (size_t)B_*T_*256*4);
    float*    ddv     = (float*)   ws_take(cur, (size_t)B_*T_*4*NH*4);
    float*    logits  = (float*)   ws_take(cur, (size_t)B_*NH*T_*S_*4);
    uint16_t* prob_bf = (uint16_t*)ws_take(cur, (size_t)B_*NH*T_*S_*2);
    float*    o_f32   = (float*)   ws_take(cur, (size_t)B_*T_*D_*4);
    uint16_t* o_bf    = (uint16_t*)ws_take(cur, (size_t)B_*T_*D_*2);

    const int BT = B_ * T_;
    auto blocks = [](size_t n) { return (unsigned)((n + 255) / 256); };

    // --- precision conversions / weight transposes ---
    f32_to_bf16_k<<<blocks((size_t)BT*D_), 256, 0, stream>>>(x, x_bf, BT*D_);
    transpose_to_bf16_k<<<blocks((size_t)D_*3*D_), 256, 0, stream>>>(w_qkv, wqkvT, D_, 3*D_);
    f32_to_bf16_k<<<blocks((size_t)D_*D_), 256, 0, stream>>>(w_o, wo_bf, D_*D_);
    transpose_to_bf16_k<<<blocks((size_t)D_*C4*KDYN), 256, 0, stream>>>(dw1, dw1T, D_, C4*KDYN);

    // --- qkv = x @ w_qkv ---
    gemm_bf16_wmma<<<dim3(3*D_/TN, BT/TM, 1), 256, 0, stream>>>(
        x_bf, 0, 0, D_,  wqkvT, 0, 0, D_,  qkvf, 0, 0, 3*D_,  BT, 3*D_, D_, 1);
    split_qkv_k<<<blocks((size_t)BT*D_), 256, 0, stream>>>(qkvf, q_bf, k_bf, vT_bf);

    // --- dynamic weights: dw_hidden = gelu(x @ dw1); w; dd ---
    gemm_bf16_wmma<<<dim3(C4*KDYN/TN, BT/TM, 1), 256, 0, stream>>>(
        x_bf, 0, 0, D_,  dw1T, 0, 0, D_,  dwh, 0, 0, C4*KDYN,  BT, C4*KDYN, D_, 1);
    gelu_k<<<blocks((size_t)BT*C4*KDYN), 256, 0, stream>>>(dwh, BT*C4*KDYN);
    dyn_w_proj_k<<<BT, 256, 0, stream>>>(dwh, qkw, w_all);
    dyn_dd_k<<<blocks((size_t)BT*4*NH), 256, 0, stream>>>(x, dd_w, ddv);

    // --- logits = q @ k^T  (batched over B*NH) ---
    gemm_bf16_wmma<<<dim3(S_/TN, T_/TM, B_*NH), 256, 0, stream>>>(
        q_bf, (long long)NH*T_*HD_, (long long)T_*HD_, HD_,
        k_bf, (long long)NH*T_*HD_, (long long)T_*HD_, HD_,
        logits, (long long)NH*T_*S_, (long long)T_*S_, S_,
        T_, S_, HD_, NH);

    // --- pre cross-head proj + causal mask, softmax, post proj ---
    cross_head_proj_k<false><<<dim3(S_/256, T_, B_), 256, 0, stream>>>(logits, w_all, ddv, nullptr);
    softmax_rows_k<<<(B_*NH*T_)/8, 256, 0, stream>>>(logits, B_*NH*T_);
    cross_head_proj_k<true ><<<dim3(S_/256, T_, B_), 256, 0, stream>>>(logits, w_all, ddv, prob_bf);

    // --- o = probs @ v  (B = vT[(b,n),HD,S]); C written head-interleaved into [B,T,D] ---
    gemm_bf16_wmma<<<dim3(HD_/TN, T_/TM, B_*NH), 256, 0, stream>>>(
        prob_bf, (long long)NH*T_*S_, (long long)T_*S_, S_,
        vT_bf,   (long long)NH*HD_*T_, (long long)HD_*T_, T_,
        o_f32,   (long long)T_*D_,     (long long)HD_,    D_,
        T_, HD_, S_, NH);

    // --- out = o @ w_o^T ---
    f32_to_bf16_k<<<blocks((size_t)BT*D_), 256, 0, stream>>>(o_f32, o_bf, BT*D_);
    gemm_bf16_wmma<<<dim3(D_/TN, BT/TM, 1), 256, 0, stream>>>(
        o_bf, 0, 0, D_,  wo_bf, 0, 0, D_,  (float*)d_out, 0, 0, D_,  BT, D_, D_, 1);
}
